// VQVAE_69664369541364
// MI455X (gfx1250) — matrix operand order repaired and checked by reading, not compiled, not run on hardware
//
#include <hip/hip_runtime.h>
#include <hip/hip_bf16.h>

// ---------------------------------------------------------------------------
// Types / helpers
// ---------------------------------------------------------------------------
typedef unsigned short u16;
typedef __attribute__((ext_vector_type(16))) __bf16 v16bf;
typedef __attribute__((ext_vector_type(8)))  float  v8f;
typedef int v4i __attribute__((vector_size(16)));   // matches builtin pointee type

__device__ __forceinline__ u16 f2bf(float f) {
  unsigned u = __float_as_uint(f);
  u += 0x7FFFu + ((u >> 16) & 1u);          // round-to-nearest-even
  return (u16)(u >> 16);
}
__device__ __forceinline__ float bf2f(u16 h) {
  return __uint_as_float(((unsigned)h) << 16);
}

// ---- gfx1250 async global->LDS copies (ASYNCcnt-tracked), with fallback ----
#if defined(__gfx1250__) && __has_builtin(__builtin_amdgcn_global_load_async_to_lds_b128)
#define HAVE_ASYNC_LDS 1
#define ASYNC_B128(g, l)                                                       \
  __builtin_amdgcn_global_load_async_to_lds_b128(                              \
      (__attribute__((address_space(1))) v4i*)(g),                             \
      (__attribute__((address_space(3))) v4i*)(l), 0, 0)
#if __has_builtin(__builtin_amdgcn_s_wait_asynccnt)
#define WAIT_ASYNC() __builtin_amdgcn_s_wait_asynccnt(0)
#else
#define WAIT_ASYNC() asm volatile("s_wait_asynccnt 0x0" ::: "memory")
#endif
#else
#define HAVE_ASYNC_LDS 0
#define ASYNC_B128(g, l) (*(uint4*)(l) = *(const uint4*)(g))
#define WAIT_ASYNC()
#endif

// ---------------------------------------------------------------------------
// Implicit-GEMM convolution with WMMA bf16 (fused bias/BN/residual/activation)
//   M = N*Ho*Wo (output pixels), N = Cout, K = R*R*Cin (padded to 32)
//   Tile: 128(M) x 64(N), KB=32, 256 threads = 8 waves, each wave 32x32
//   Double-buffered LDS tiles filled with async global->LDS b128 copies.
//   Fast A path when Cin%16==0: one im2col tap = two b128 transfers.
//   dilated==1 -> ConvTranspose2d(k=4,s=2,p=1) as lhs-dilated conv (pad=2)
//   act: 0=none 1=relu 2=sigmoid
// ---------------------------------------------------------------------------
#define TILE_M 128
#define TILE_N 64
#define KB 32

__global__ __launch_bounds__(256)
void k_conv_wmma(const u16* __restrict__ in, int Nn, int H, int W, int Cin,
                 const u16* __restrict__ wT, int Kpad, int Ktot,
                 const float* __restrict__ scale, const float* __restrict__ shift,
                 const u16* __restrict__ resid,
                 u16* __restrict__ out_bf, float* __restrict__ out_f32,
                 float* __restrict__ out_nchw,
                 int Cout, int Ho, int Wo,
                 int R, int stride, int pad, int dilated, int act)
{
  __shared__ __align__(16) u16 As[2][TILE_M][KB];
  __shared__ __align__(16) u16 Bs[2][TILE_N][KB];

  const int tid  = threadIdx.x;
  const int lane = tid & 31;
  const int wave = tid >> 5;        // 0..7
  const int wm   = wave >> 1;       // 0..3 (M quarter)
  const int wn   = wave & 1;        // 0..1 (N half)

  const int M     = Nn * Ho * Wo;
  const int mBase = blockIdx.x * TILE_M;
  const int nBase = blockIdx.y * TILE_N;

  v8f acc[2][2] = {};

  // A-tile fill assignment: row = tid>>1, 16 contiguous k per thread
  const int arow = tid >> 1;            // 0..127
  const int ah   = (tid & 1) * 16;      // 0 or 16
  const int p    = mBase + arow;
  const bool prow = p < M;
  int pn = 0, py = 0, px = 0;
  if (prow) {
    pn = p / (Ho * Wo);
    int rp = p - pn * Ho * Wo;
    py = rp / Wo;
    px = rp - py * Wo;
  }
  const size_t inBase = (size_t)pn * H * W * Cin;
  const bool fastA = ((Cin & 15) == 0);   // all layers except the 3-ch stem

  // B-tile fill assignment: row = tid>>2, 8 contiguous k per thread
  const int brow = tid >> 2;            // 0..63
  const int bk   = (tid & 3) * 8;       // 0,8,16,24

  auto fill = [&](int buf, int kt) {
    // ---- B tile (weights pre-packed [Cout][Kpad] bf16): contiguous -> async
    {
      int n = nBase + brow;
      if (n < Cout) {
        ASYNC_B128(wT + (size_t)n * Kpad + kt + bk, &Bs[buf][brow][bk]);
      } else {
        uint4 z = {0u, 0u, 0u, 0u};
        *(uint4*)&Bs[buf][brow][bk] = z;
      }
    }
    // ---- A tile: implicit im2col gather
    if (fastA) {
      // 16 consecutive k are one (ky,kx) tap, contiguous 32B in NHWC memory.
      // (Kpad==Ktot for every Cin%16==0 layer, so no k<Ktot masking needed.)
      const int k0  = kt + ah;
      const int rem = k0 / Cin;
      const int ci  = k0 - rem * Cin;
      const int kx  = rem % R;
      const int ky  = rem / R;
      bool ok = prow;
      int iy = 0, ix = 0;
      if (ok) {
        if (!dilated) {
          iy = py * stride - pad + ky;
          ix = px * stride - pad + kx;
          ok = (unsigned)iy < (unsigned)H && (unsigned)ix < (unsigned)W;
        } else {
          int ty = py + ky - pad, tx = px + kx - pad;
          ok = (ty >= 0) && (tx >= 0) && !(ty & 1) && !(tx & 1);
          iy = ty >> 1;
          ix = tx >> 1;
          ok = ok && (iy < H) && (ix < W);
        }
      }
      if (ok) {
        const u16* g = in + inBase + (size_t)(iy * W + ix) * Cin + ci;
        ASYNC_B128(g,     &As[buf][arow][ah]);
        ASYNC_B128(g + 8, &As[buf][arow][ah + 8]);
      } else {
        uint4 z = {0u, 0u, 0u, 0u};
        *(uint4*)&As[buf][arow][ah]     = z;
        *(uint4*)&As[buf][arow][ah + 8] = z;
      }
    } else {
      // scalar path (stem: Cin=3, Ktot=48 < Kpad=64)
      u16 vals[16];
      int k0  = kt + ah;
      int ci  = k0 % Cin;
      int rem = k0 / Cin;
      int kx  = rem % R;
      int ky  = rem / R;
      #pragma unroll
      for (int j = 0; j < 16; ++j) {
        int k = k0 + j;
        u16 v = 0;
        if (prow && k < Ktot) {
          int iy, ix;
          bool ok;
          if (!dilated) {
            iy = py * stride - pad + ky;
            ix = px * stride - pad + kx;
            ok = (unsigned)iy < (unsigned)H && (unsigned)ix < (unsigned)W;
          } else {
            int ty = py + ky - pad, tx = px + kx - pad;
            ok = (ty >= 0) && (tx >= 0) && !(ty & 1) && !(tx & 1);
            iy = ty >> 1;
            ix = tx >> 1;
            ok = ok && (iy < H) && (ix < W);
          }
          if (ok) v = in[inBase + (size_t)(iy * W + ix) * Cin + ci];
        }
        vals[j] = v;
        if (++ci == Cin) { ci = 0; if (++kx == R) { kx = 0; ++ky; } }
      }
      #pragma unroll
      for (int j = 0; j < 16; ++j) As[buf][arow][ah + j] = vals[j];
    }
  };

  // ---- software-pipelined main loop (double-buffered LDS)
  fill(0, 0);
  WAIT_ASYNC();
  __syncthreads();

  int buf = 0;
  for (int kt = 0; kt < Kpad; kt += KB) {
    if (kt + KB < Kpad) fill(buf ^ 1, kt + KB);   // prefetch next tile (async)

    // ---- fragments (CDNA5 16-bit A 16x32 / B 32x16 VGPR layouts, ISA 7.12.2)
    union FragBF { v16bf v; uint4 q[2]; };
    FragBF a[2], b[2];
    {
      const int mrow = wm * 32 + (lane & 15);
      const int kb   = (lane < 16) ? 0 : 8;       // A: K halves split by lane group
      a[0].q[0] = *(const uint4*)&As[buf][mrow][kb];
      a[0].q[1] = *(const uint4*)&As[buf][mrow][kb + 16];
      a[1].q[0] = *(const uint4*)&As[buf][mrow + 16][kb];
      a[1].q[1] = *(const uint4*)&As[buf][mrow + 16][kb + 16];
      const int nrow = wn * 32 + (lane & 15);
      const int kb2  = (lane < 16) ? 0 : 8;       // B: K halves split by lane group (same as A)
      b[0].q[0] = *(const uint4*)&Bs[buf][nrow][kb2];
      b[0].q[1] = *(const uint4*)&Bs[buf][nrow][kb2 + 16];
      b[1].q[0] = *(const uint4*)&Bs[buf][nrow + 16][kb2];
      b[1].q[1] = *(const uint4*)&Bs[buf][nrow + 16][kb2 + 16];
    }
    #pragma unroll
    for (int r = 0; r < 2; ++r)
      #pragma unroll
      for (int c = 0; c < 2; ++c)
        acc[r][c] = __builtin_amdgcn_wmma_f32_16x16x32_bf16(
            false, a[r].v, false, b[c].v, (short)0, acc[r][c], false, false);

    WAIT_ASYNC();        // async fills of buf^1 done before barrier
    __syncthreads();
    buf ^= 1;
  }

  // ---- epilogue: scale/shift (folded BN+bias), residual, activation, store
  #pragma unroll
  for (int r = 0; r < 2; ++r) {
    #pragma unroll
    for (int c = 0; c < 2; ++c) {
      const int n = nBase + wn * 32 + c * 16 + (lane & 15);
      if (n >= Cout) continue;
      const float sc = scale ? scale[n] : 1.0f;
      const float sh = shift ? shift[n] : 0.0f;
      #pragma unroll
      for (int i = 0; i < 8; ++i) {
        const int m = mBase + wm * 32 + r * 16 + ((lane < 16) ? i : i + 8);
        if (m >= M) continue;
        float o = acc[r][c][i] * sc + sh;
        if (resid) o += bf2f(resid[(size_t)m * Cout + n]);
        if (act == 1)      o = fmaxf(o, 0.0f);
        else if (act == 2) o = 1.0f / (1.0f + __expf(-o));
        if (out_bf)  out_bf[(size_t)m * Cout + n]  = f2bf(o);
        if (out_f32) out_f32[(size_t)m * Cout + n] = o;
        if (out_nchw) {
          const int nb = m / (Ho * Wo);
          const int rp = m - nb * Ho * Wo;
          const int oy = rp / Wo, ox = rp - oy * Wo;
          out_nchw[((size_t)(nb * Cout + n) * Ho + oy) * Wo + ox] = o;
        }
      }
    }
  }
}

// ---------------------------------------------------------------------------
// Prep kernels
// ---------------------------------------------------------------------------
__global__ void k_x_cvt(const float* __restrict__ x, u16* __restrict__ out,
                        int N, int C, int H, int W)
{
  size_t i = (size_t)blockIdx.x * blockDim.x + threadIdx.x;
  size_t total = (size_t)N * C * H * W;
  if (i >= total) return;
  int c = (int)(i % C);
  size_t r = i / C;
  int xw = (int)(r % W); r /= W;
  int y  = (int)(r % H);
  int n  = (int)(r / H);
  out[i] = f2bf(x[(((size_t)n * C + c) * H + y) * W + xw]);
}

// OIHW fp32 -> [Cout][Kpad] bf16, k = (ky*R+kx)*Cin + ci
__global__ void k_pack_conv_w(const float* __restrict__ w, u16* __restrict__ dst,
                              int Cout, int Cin, int R, int Kpad, int K)
{
  int i = blockIdx.x * blockDim.x + threadIdx.x;
  if (i >= Cout * Kpad) return;
  int co = i / Kpad, k = i - co * Kpad;
  u16 v = 0;
  if (k < K) {
    int ci = k % Cin, rem = k / Cin;
    int kx = rem % R, ky = rem / R;
    v = f2bf(w[(((size_t)co * Cin + ci) * R + ky) * R + kx]);
  }
  dst[i] = v;
}

// ConvT weight (Cin,Cout,4,4) -> flipped/transposed [Cout][Kpad] bf16
__global__ void k_pack_convT_w(const float* __restrict__ w, u16* __restrict__ dst,
                               int Cout, int Cin, int Kpad, int K)
{
  int i = blockIdx.x * blockDim.x + threadIdx.x;
  if (i >= Cout * Kpad) return;
  int co = i / Kpad, k = i - co * Kpad;
  u16 v = 0;
  if (k < K) {
    int ci = k % Cin, rem = k / Cin;
    int kx = rem & 3, ky = rem >> 2;
    v = f2bf(w[(((size_t)ci * Cout + co) * 4 + (3 - ky)) * 4 + (3 - kx)]);
  }
  dst[i] = v;
}

// BN fold: scale = g*rsqrt(v+eps); shift = (b-m)*scale + be.  g==null -> bias only
__global__ void k_fold_bn(const float* __restrict__ b, const float* __restrict__ g,
                          const float* __restrict__ be, const float* __restrict__ m,
                          const float* __restrict__ v,
                          float* __restrict__ sc, float* __restrict__ sh, int C)
{
  int c = blockIdx.x * blockDim.x + threadIdx.x;
  if (c >= C) return;
  if (g) {
    float s = g[c] * rsqrtf(v[c] + 1e-5f);
    sc[c] = s;
    sh[c] = (b[c] - m[c]) * s + be[c];
  } else {
    sc[c] = 1.0f;
    sh[c] = b[c];
  }
}

// ---------------------------------------------------------------------------
// Vector quantization
// ---------------------------------------------------------------------------
__global__ void k_vq_zero(float* loss_acc, float* counts)
{
  int i = blockIdx.x * blockDim.x + threadIdx.x;
  if (i == 0) *loss_acc = 0.0f;
  if (i < 512) counts[i] = 0.0f;
}

__global__ void k_vq(const float* __restrict__ z, const float* __restrict__ emb,
                     u16* __restrict__ q, float* __restrict__ idx_out,
                     float* __restrict__ loss_acc, float* __restrict__ counts, int P)
{
  int p = blockIdx.x * blockDim.x + threadIdx.x;
  if (p >= P) return;
  float zv[64];
  #pragma unroll
  for (int d = 0; d < 64; ++d) zv[d] = z[(size_t)p * 64 + d];
  int best = 0;
  float bd = 3.4e38f;
  for (int k = 0; k < 512; ++k) {
    const float* e = emb + (size_t)k * 64;
    float s = 0.0f;
    #pragma unroll
    for (int d = 0; d < 64; ++d) { float df = zv[d] - e[d]; s = fmaf(df, df, s); }
    if (s < bd) { bd = s; best = k; }
  }
  idx_out[p] = (float)best;
  const float* e = emb + (size_t)best * 64;
  float se = 0.0f;
  #pragma unroll
  for (int d = 0; d < 64; ++d) {
    q[(size_t)p * 64 + d] = f2bf(e[d]);
    float df = e[d] - zv[d];
    se = fmaf(df, df, se);
  }
  atomicAdd(loss_acc, se);
  atomicAdd(&counts[best], 1.0f);
}

__global__ void k_vq_fin(const float* __restrict__ loss_acc, const float* __restrict__ counts,
                         float* __restrict__ out_loss, float* __restrict__ out_perp,
                         int P, float cc)
{
  __shared__ float red[256];
  float s = 0.0f;
  for (int k = threadIdx.x; k < 512; k += 256) {
    float avg = counts[k] * (1.0f / (float)P);
    s += avg * __logf(avg + 1e-10f);
  }
  red[threadIdx.x] = s;
  __syncthreads();
  for (int o = 128; o > 0; o >>= 1) {
    if ((int)threadIdx.x < o) red[threadIdx.x] += red[threadIdx.x + o];
    __syncthreads();
  }
  if (threadIdx.x == 0) {
    *out_perp = __expf(-red[0]);
    *out_loss = (*loss_acc) * (cc / (float)((size_t)P * 64));
  }
}

// ---------------------------------------------------------------------------
// Host orchestration
// ---------------------------------------------------------------------------
extern "C" void kernel_launch(void* const* d_in, const int* in_sizes, int n_in,
                              void* d_out, int out_size, void* d_ws, size_t ws_size,
                              hipStream_t stream)
{
  (void)in_sizes; (void)n_in; (void)out_size; (void)ws_size;
  #define IN_F(i) ((const float*)d_in[(i)])

  // d_in flattening assumption: dict insertion order {x, params, embedding};
  // inner dicts flattened with sorted keys (jax tree order); lists in order.
  //   0: x
  //   params.dec: out_c{b=1,w=2}, res1[rb0@3, rb1@15], res2[rb0@27, rb1@39],
  //               stem_c{b=51,w=52}, stem_n@53, up1_c{b=57,w=58}, up1_n@59,
  //               up2_c{b=63,w=64}, up2_n@65
  //   params.enc: down_c{b=69,w=70}, down_n@71, head_c{b=75,w=76}, head_n@77,
  //               res1[rb0@81, rb1@93], res2[rb0@105, rb1@117],
  //               stem_c{b=129,w=130}, stem_n@131
  //   135: embedding
  // resblock leaf order: c1.b,c1.w,c2.b,c2.w, n1.{be,g,m,v}, n2.{be,g,m,v}

  const int B = 32;
  char* wsp = (char*)d_ws;
  size_t off = 0;
  auto alloc = [&](size_t bytes) -> void* {
    void* pp = wsp + off;
    off = (off + bytes + 255) & ~(size_t)255;
    return pp;
  };

  // activation buffers (max: 32*128*128*64 bf16)
  const size_t actElems = (size_t)B * 128 * 128 * 64;
  u16* bufA = (u16*)alloc(actElems * 2);
  u16* bufB = (u16*)alloc(actElems * 2);
  u16* bufC = (u16*)alloc(actElems * 2);
  u16* bufX = (u16*)alloc((size_t)B * 128 * 128 * 3 * 2);
  float* zbuf = (float*)alloc((size_t)B * 32 * 32 * 64 * 4);
  u16*   qbuf = (u16*)alloc((size_t)B * 32 * 32 * 64 * 2);
  float* vq_loss = (float*)alloc(4);
  float* vq_cnt  = (float*)alloc(512 * 4);

  // ---- prep lambdas
  auto packConv = [&](int wIdx, int Cout, int Cin, int R) -> u16* {
    int K = R * R * Cin, Kp = (K + 31) & ~31;
    u16* dst = (u16*)alloc((size_t)Cout * Kp * 2);
    int total = Cout * Kp;
    k_pack_conv_w<<<(total + 255) / 256, 256, 0, stream>>>(IN_F(wIdx), dst, Cout, Cin, R, Kp, K);
    return dst;
  };
  auto packConvT = [&](int wIdx, int Cout, int Cin) -> u16* {
    int K = 16 * Cin, Kp = (K + 31) & ~31;
    u16* dst = (u16*)alloc((size_t)Cout * Kp * 2);
    int total = Cout * Kp;
    k_pack_convT_w<<<(total + 255) / 256, 256, 0, stream>>>(IN_F(wIdx), dst, Cout, Cin, Kp, K);
    return dst;
  };
  auto foldBN = [&](int bIdx, int bnBase, int C, float** sc, float** sh) {
    *sc = (float*)alloc((size_t)C * 4);
    *sh = (float*)alloc((size_t)C * 4);
    const float* g  = bnBase >= 0 ? IN_F(bnBase + 1) : nullptr;
    const float* be = bnBase >= 0 ? IN_F(bnBase + 0) : nullptr;
    const float* m  = bnBase >= 0 ? IN_F(bnBase + 2) : nullptr;
    const float* v  = bnBase >= 0 ? IN_F(bnBase + 3) : nullptr;
    k_fold_bn<<<(C + 255) / 256, 256, 0, stream>>>(IN_F(bIdx), g, be, m, v, *sc, *sh, C);
  };
  auto conv = [&](const u16* in, int H, int W, int Cin, const u16* wT, int Kpad, int Ktot,
                  const float* sc, const float* sh, const u16* resid,
                  u16* ob, float* of, float* onchw,
                  int Cout, int Ho, int Wo, int R, int st, int pd, int dil, int act) {
    dim3 grid((unsigned)((B * Ho * Wo + TILE_M - 1) / TILE_M),
              (unsigned)((Cout + TILE_N - 1) / TILE_N));
    k_conv_wmma<<<grid, 256, 0, stream>>>(in, B, H, W, Cin, wT, Kpad, Ktot, sc, sh, resid,
                                          ob, of, onchw, Cout, Ho, Wo, R, st, pd, dil, act);
  };

  // ---- pack all weights / fold all BN (recomputed every call; deterministic)
  u16 *w_rb[8][2]; float *sc_rb[8][2], *sh_rb[8][2];
  const int rbBase[8] = {81, 93, 105, 117, 3, 15, 27, 39}; // enc.res1 x2, enc.res2 x2, dec.res1 x2, dec.res2 x2
  const int rbCh[8]   = {128, 128, 256, 256, 256, 256, 128, 128};
  for (int r = 0; r < 8; ++r) {
    int base = rbBase[r], C = rbCh[r];
    w_rb[r][0] = packConv(base + 1, C, C, 3);
    w_rb[r][1] = packConv(base + 3, C, C, 3);
    foldBN(base + 0, base + 4, C, &sc_rb[r][0], &sh_rb[r][0]);
    foldBN(base + 2, base + 8, C, &sc_rb[r][1], &sh_rb[r][1]);
  }
  u16* w_stem  = packConv(130, 128, 3, 4);   float *sc_stem, *sh_stem;  foldBN(129, 131, 128, &sc_stem, &sh_stem);
  u16* w_down  = packConv(70, 256, 128, 4);  float *sc_down, *sh_down;  foldBN(69, 71, 256, &sc_down, &sh_down);
  u16* w_head  = packConv(76, 64, 256, 1);   float *sc_head, *sh_head;  foldBN(75, 77, 64, &sc_head, &sh_head);
  u16* w_dstem = packConv(52, 256, 64, 1);   float *sc_dstem,*sh_dstem; foldBN(51, 53, 256, &sc_dstem, &sh_dstem);
  u16* w_up1   = packConvT(58, 128, 256);    float *sc_up1, *sh_up1;    foldBN(57, 59, 128, &sc_up1, &sh_up1);
  u16* w_up2   = packConvT(64, 64, 128);     float *sc_up2, *sh_up2;    foldBN(63, 65, 64, &sc_up2, &sh_up2);
  u16* w_out   = packConv(2, 3, 64, 3);      float *sc_out, *sh_out;    foldBN(1, -1, 3, &sc_out, &sh_out);

  auto resblock = [&](int r, u16* xin, u16* t, u16* outb, int H, int W) {
    int C = rbCh[r], Kp = 9 * C;
    conv(xin, H, W, C, w_rb[r][0], Kp, Kp, sc_rb[r][0], sh_rb[r][0], nullptr,
         t, nullptr, nullptr, C, H, W, 3, 1, 1, 0, 1);
    conv(t, H, W, C, w_rb[r][1], Kp, Kp, sc_rb[r][1], sh_rb[r][1], xin,
         outb, nullptr, nullptr, C, H, W, 3, 1, 1, 0, 1);
  };

  // ---- input: NCHW fp32 -> NHWC bf16
  {
    size_t total = (size_t)B * 3 * 128 * 128;
    k_x_cvt<<<(unsigned)((total + 255) / 256), 256, 0, stream>>>(IN_F(0), bufX, B, 3, 128, 128);
  }

  // ---- encoder
  conv(bufX, 128, 128, 3, w_stem, 64, 48, sc_stem, sh_stem, nullptr,
       bufA, nullptr, nullptr, 128, 64, 64, 4, 2, 1, 0, 1);
  resblock(0, bufA, bufB, bufC, 64, 64);
  resblock(1, bufC, bufA, bufB, 64, 64);
  conv(bufB, 64, 64, 128, w_down, 2048, 2048, sc_down, sh_down, nullptr,
       bufA, nullptr, nullptr, 256, 32, 32, 4, 2, 1, 0, 1);
  resblock(2, bufA, bufB, bufC, 32, 32);
  resblock(3, bufC, bufA, bufB, 32, 32);
  conv(bufB, 32, 32, 256, w_head, 256, 256, sc_head, sh_head, nullptr,
       nullptr, zbuf, nullptr, 64, 32, 32, 1, 1, 0, 0, 0);

  // ---- VQ
  float* outf = (float*)d_out;
  const size_t RECON = (size_t)B * 3 * 128 * 128; // 1,572,864
  float* loss_out = outf + RECON;
  float* idx_out  = outf + RECON + 1;
  float* perp_out = outf + RECON + 1 + (size_t)B * 32 * 32;
  const int P = B * 32 * 32;
  k_vq_zero<<<2, 256, 0, stream>>>(vq_loss, vq_cnt);
  k_vq<<<(P + 255) / 256, 256, 0, stream>>>(zbuf, IN_F(135), qbuf, idx_out, vq_loss, vq_cnt, P);
  k_vq_fin<<<1, 256, 0, stream>>>(vq_loss, vq_cnt, loss_out, perp_out, P, 0.25f);

  // ---- decoder
  conv(qbuf, 32, 32, 64, w_dstem, 64, 64, sc_dstem, sh_dstem, nullptr,
       bufA, nullptr, nullptr, 256, 32, 32, 1, 1, 0, 0, 1);
  resblock(4, bufA, bufB, bufC, 32, 32);
  resblock(5, bufC, bufA, bufB, 32, 32);
  conv(bufB, 32, 32, 256, w_up1, 4096, 4096, sc_up1, sh_up1, nullptr,
       bufA, nullptr, nullptr, 128, 64, 64, 4, 1, 2, 1, 1);
  resblock(6, bufA, bufB, bufC, 64, 64);
  resblock(7, bufC, bufA, bufB, 64, 64);
  conv(bufB, 64, 64, 128, w_up2, 2048, 2048, sc_up2, sh_up2, nullptr,
       bufA, nullptr, nullptr, 64, 128, 128, 4, 1, 2, 1, 1);
  conv(bufA, 128, 128, 64, w_out, 576, 576, sc_out, sh_out, nullptr,
       nullptr, nullptr, outf, 3, 128, 128, 3, 1, 1, 0, 2);

  #undef IN_F
}
